// MambaModel_19688130085062
// MI455X (gfx1250) — matrix operand order repaired and checked by reading, not compiled, not run on hardware
//
#include <hip/hip_runtime.h>

// ---- model constants (match reference) ----
#define DM      256
#define DI      512      // D_INNER
#define DTR     16       // DT_RANK
#define DSTATE  16
#define NB      8        // batch
#define LSEQ    4096
#define NLAYERS 4
#define ROWS    (NB * LSEQ)   // 32768 GEMM rows

typedef __attribute__((ext_vector_type(16))) __bf16 v16bf;
typedef __attribute__((ext_vector_type(8)))  __bf16 v8bf;
typedef __attribute__((ext_vector_type(8)))  float  v8f;
typedef __attribute__((ext_vector_type(4)))  unsigned int u32x4;
typedef __attribute__((ext_vector_type(8)))  int i32x8;
typedef __attribute__((ext_vector_type(4)))  int i32x4;

#ifndef __has_builtin
#define __has_builtin(x) 0
#endif
#if __has_builtin(__builtin_amdgcn_tensor_load_to_lds) && \
    __has_builtin(__builtin_amdgcn_s_wait_tensorcnt)
#define HAS_TDM 1
#else
#define HAS_TDM 0
#endif

// ---------------------------------------------------------------------------
// TDM: async-load a contiguous strip of `rows` rows x K bf16 (row stride K)
// from global into LDS. Issued by the executing wave; completion via TENSORcnt.
// D# layout per CDNA5 ISA 8.3/8.4 (group0: flags/lds/global addr, group1:
// dims/tile/strides). 2D tile: tile_dim0 = K elems, tile_dim1 = rows.
// ---------------------------------------------------------------------------
__device__ __forceinline__ void tdm_load_strip(const __bf16* gsrc, void* lds_dst,
                                               int K, int rows, int M)
{
#if HAS_TDM
    unsigned long long ga = (unsigned long long)(const void*)gsrc;
    unsigned lds_off = (unsigned)(unsigned long long)lds_dst; // low 32 = LDS byte offset
    u32x4 g0;
    g0[0] = 1u;                                        // count=1, user mode
    g0[1] = lds_off;                                   // lds_addr
    g0[2] = (unsigned)(ga & 0xFFFFFFFFu);              // global_addr[31:0]
    g0[3] = (unsigned)((ga >> 32) & 0x1FFFFFFu) | (2u << 30); // addr[56:32] | type=2
    i32x8 g1;
    g1[0] = 0x00010000;                                // wg_mask=0, data_size=1 (2B)
    g1[1] = (int)((K & 0xFFFF) << 16);                 // tensor_dim0[15:0]
    g1[2] = (int)(((unsigned)M & 0xFFFFu) << 16);      // tensor_dim0[31:16]=0 | tensor_dim1[15:0]
    g1[3] = (int)((((unsigned)M >> 16) & 0xFFFFu) | ((unsigned)(K & 0xFFFF) << 16)); // dim1 hi | tile_dim0
    g1[4] = rows;                                      // tile_dim1 | tile_dim2=0
    g1[5] = K;                                         // tensor_dim0_stride[31:0]
    g1[6] = 0;
    g1[7] = 0;
    i32x4 g2 = {0, 0, 0, 0};
    i32x4 g3 = {0, 0, 0, 0};
#if __clang_major__ >= 23
    i32x8 g4 = {0, 0, 0, 0, 0, 0, 0, 0};
    __builtin_amdgcn_tensor_load_to_lds(g0, g1, g2, g3, g4, 0);
#else
    __builtin_amdgcn_tensor_load_to_lds(g0, g1, g2, g3, 0);
#endif
    __builtin_amdgcn_s_wait_tensorcnt(0);
#else
    (void)gsrc; (void)lds_dst; (void)K; (void)rows; (void)M;
#endif
}

// ---------------------------------------------------------------------------
// bf16 WMMA GEMM:  C[M,N] = A[M,K] @ W[N,K]^T   (W row-major [N,K] == B^T)
// One wave computes a 16 x (16*NT) tile; f32 accumulation.
// N = CW*16*NT (compile time). When LDSA: the block's A strip is staged in
// LDS via TDM (waves sharing row-tiles read it with ds_load_b128).
// Fragment layouts per CDNA5 ISA 7.12.2 (wave32).
// ---------------------------------------------------------------------------
template <int NT, int CW, int K, bool LDSA>
__global__ __launch_bounds__(256) void gemm_bf16_wmma(
    const __bf16* __restrict__ A, const __bf16* __restrict__ W,
    float* __restrict__ Cf, __bf16* __restrict__ Cb, int M)
{
    constexpr int N   = CW * 16 * NT;
    constexpr int CPB = (CW < 8) ? CW : 8;   // col tiles covered by one block
    constexpr int RPB = 8 / CPB;             // row tiles per block
    constexpr int ROWSPB = RPB * 16;
    __shared__ __bf16 sA[LDSA ? (ROWSPB * K) : 16];

    const int lane = threadIdx.x & 31;
    const int w    = threadIdx.x >> 5;
    const int grp  = lane >> 4;
    const int l16  = lane & 15;

    constexpr int BPRG = CW / CPB;           // blocks covering one row group
    const int rowTile0 = (blockIdx.x / BPRG) * RPB;
    const int localRT  = w / CPB;
    const int rowTile  = rowTile0 + localRT;
    const int colTile  = (blockIdx.x % BPRG) * CPB + (w % CPB);
    const int n0       = colTile * 16 * NT;

    const __bf16* Arow;                       // base of this lane's A row
    int rstride;                              // elements between A rows
    if (LDSA) {
        if (threadIdx.x < 32) {
            tdm_load_strip(A + (size_t)rowTile0 * 16 * K, (void*)sA, K, ROWSPB, M);
        }
        __syncthreads();
        Arow = sA + (localRT * 16 + l16) * K;
        rstride = K;
    } else {
        Arow = A + (size_t)(rowTile * 16 + l16) * K;
        rstride = K;
    }
    (void)rstride;

    v8f acc[NT] = {};

    for (int k0 = 0; k0 < K; k0 += 32) {
        // A fragment (16x32): lanes 0-15 hold K {0..7,16..23}, lanes 16-31 {8..15,24..31}
        const __bf16* pa = Arow + k0 + grp * 8;
        v8bf alo = *(const v8bf*)pa;
        v8bf ahi = *(const v8bf*)(pa + 16);
        v16bf afrag = __builtin_shufflevector(alo, ahi,
                          0,1,2,3,4,5,6,7,8,9,10,11,12,13,14,15);
        // prefetch next k-step of the weight stream into near cache
        if (k0 + 32 < K)
            __builtin_prefetch((const void*)(W + (size_t)(n0 + l16) * K + k0 + 32), 0, 3);
#pragma unroll
        for (int t = 0; t < NT; ++t) {
            // B fragment (32x16): lane holds column n, K {grp*8..+7, 16+grp*8..+7}
            const __bf16* pb = W + (size_t)(n0 + t * 16 + l16) * K + k0 + grp * 8;
            v8bf blo = *(const v8bf*)pb;
            v8bf bhi = *(const v8bf*)(pb + 16);
            v16bf bfrag = __builtin_shufflevector(blo, bhi,
                              0,1,2,3,4,5,6,7,8,9,10,11,12,13,14,15);
            acc[t] = __builtin_amdgcn_wmma_f32_16x16x32_bf16(
                         false, afrag, false, bfrag,
                         (short)0, acc[t], false, false);
        }
    }

    // D layout: VGPR r -> row (r + 8*grp), col = lane&15
#pragma unroll
    for (int t = 0; t < NT; ++t) {
        const int nn = n0 + t * 16 + l16;
#pragma unroll
        for (int r = 0; r < 8; ++r) {
            const int mm = rowTile * 16 + r + 8 * grp;
            const float v = acc[t][r];
            if (Cf) Cf[(size_t)mm * N + nn] = v;
            if (Cb) Cb[(size_t)mm * N + nn] = (__bf16)v;
        }
    }
}

// ---------------------------------------------------------------------------
// x = condition + pe, stored as bf16 for the in_proj WMMA
// ---------------------------------------------------------------------------
__global__ void prep_kernel(const float* __restrict__ cond,
                            const float* __restrict__ pe,
                            __bf16* __restrict__ xb)
{
    size_t i = (size_t)blockIdx.x * blockDim.x + threadIdx.x;
    if (i >= (size_t)ROWS * DM) return;
    size_t ld = i % ((size_t)LSEQ * DM);   // l*DM + d
    xb[i] = (__bf16)(cond[i] + pe[ld]);
}

__global__ void f2bf_kernel(const float* __restrict__ src,
                            __bf16* __restrict__ dst, int n)
{
    int i = blockIdx.x * blockDim.x + threadIdx.x;
    if (i < n) dst[i] = (__bf16)src[i];
}

// ---------------------------------------------------------------------------
// depthwise causal conv(4) + bias + SiLU ; reads xi half of xz
// ---------------------------------------------------------------------------
__global__ void conv_silu_kernel(const float* __restrict__ xz,
                                 const float* __restrict__ cw,
                                 const float* __restrict__ cb,
                                 float* __restrict__ xc_f,
                                 __bf16* __restrict__ xc_b)
{
    size_t i = (size_t)blockIdx.x * blockDim.x + threadIdx.x;
    if (i >= (size_t)ROWS * DI) return;
    const int d   = (int)(i % DI);
    const long row = (long)(i / DI);       // b*L + t
    const int t   = (int)(row % LSEQ);
    float acc = cb[d];
#pragma unroll
    for (int k = 0; k < 4; ++k) {
        const int tt = t - 3 + k;
        if (tt >= 0)
            acc += cw[d * 4 + k] * xz[(size_t)(row + (tt - t)) * (2 * DI) + d];
    }
    const float s = acc / (1.f + __expf(-acc));   // SiLU
    xc_f[i] = s;
    xc_b[i] = (__bf16)s;
}

// ---------------------------------------------------------------------------
// delta = softplus(dt @ w_dt^T + b_dt), K = DT_RANK = 16
// ---------------------------------------------------------------------------
__global__ void delta_kernel(const float* __restrict__ dbc,
                             const float* __restrict__ wdt,
                             const float* __restrict__ bdt,
                             float* __restrict__ delta)
{
    size_t i = (size_t)blockIdx.x * blockDim.x + threadIdx.x;
    if (i >= (size_t)ROWS * DI) return;
    const int d = (int)(i % DI);
    const size_t row = i / DI;
    const float* dtv = dbc + row * 48;
    float acc = bdt[d];
#pragma unroll
    for (int r = 0; r < DTR; ++r) acc += dtv[r] * wdt[d * DTR + r];
    delta[i] = (acc > 20.f) ? acc : __logf(1.f + __expf(acc));
}

// ---------------------------------------------------------------------------
// Selective scan: one state (n) per lane, 16 lanes per channel d.
// Block = 256 threads = 16 channels of one batch. B/C/delta/x/z chunked in LDS.
// Fused with SiLU(z) gating; writes bf16 for the out_proj WMMA.
// ---------------------------------------------------------------------------
#define TCH 64
__global__ __launch_bounds__(256) void scan_kernel(
    const float* __restrict__ dbc,   const float* __restrict__ delta,
    const float* __restrict__ xc,    const float* __restrict__ xz,
    const float* __restrict__ alog,  const float* __restrict__ dskip,
    __bf16* __restrict__ ygate)
{
    __shared__ float sB[TCH][16], sC[TCH][16];
    __shared__ float sD[TCH][16], sX[TCH][16], sZ[TCH][16];

    const int b  = blockIdx.x >> 5;          // / (DI/16)
    const int d0 = (blockIdx.x & 31) * 16;
    const int g  = threadIdx.x >> 4;         // channel within block
    const int n  = threadIdx.x & 15;         // state index
    const int d  = d0 + g;

    const float Av = -__expf(alog[d * DSTATE + n]);
    const float Dv = dskip[d];
    float h = 0.f;
    const size_t rowBase = (size_t)b * LSEQ;

    for (int t0 = 0; t0 < LSEQ; t0 += TCH) {
        __syncthreads();
#pragma unroll
        for (int j = 0; j < 4; ++j) {
            const int e  = threadIdx.x + j * 256;
            const int tt = e >> 4, c = e & 15;
            const size_t row = rowBase + t0 + tt;
            sB[tt][c] = dbc[row * 48 + DTR + c];
            sC[tt][c] = dbc[row * 48 + DTR + DSTATE + c];
            const int dd = d0 + c;
            sD[tt][c] = delta[row * DI + dd];
            sX[tt][c] = xc[row * DI + dd];
            sZ[tt][c] = xz[row * (2 * DI) + DI + dd];
        }
        __syncthreads();

        for (int tt = 0; tt < TCH; ++tt) {
            const float dt = sD[tt][g];
            const float xt = sX[tt][g];
            h = __expf(dt * Av) * h + (dt * xt) * sB[tt][n];
            float p = h * sC[tt][n];
            p += __shfl_xor(p, 1, 32);
            p += __shfl_xor(p, 2, 32);
            p += __shfl_xor(p, 4, 32);
            p += __shfl_xor(p, 8, 32);
            if (n == 0) {
                const float y  = p + Dv * xt;
                const float z  = sZ[tt][g];
                const float yg = y * (z / (1.f + __expf(-z)));
                ygate[(rowBase + t0 + tt) * DI + d] = (__bf16)yg;
            }
        }
    }
}

// ---------------------------------------------------------------------------
extern "C" void kernel_launch(void* const* d_in, const int* in_sizes, int n_in,
                              void* d_out, int out_size, void* d_ws, size_t ws_size,
                              hipStream_t stream)
{
    (void)in_sizes; (void)n_in; (void)out_size; (void)ws_size;

    const float* cond   = (const float*)d_in[1];
    const float* pe     = (const float*)d_in[2];
    const float* w_in   = (const float*)d_in[3];
    const float* conv_w = (const float*)d_in[4];
    const float* conv_b = (const float*)d_in[5];
    const float* w_xp   = (const float*)d_in[6];
    const float* w_dt   = (const float*)d_in[7];
    const float* b_dt   = (const float*)d_in[8];
    const float* A_log  = (const float*)d_in[9];
    const float* D_skip = (const float*)d_in[10];
    const float* w_out  = (const float*)d_in[11];
    float* out = (float*)d_out;

    // workspace carve-up (256B aligned)
    char* base = (char*)d_ws;
    size_t off = 0;
    auto carve = [&](size_t bytes) {
        void* p = base + off;
        off = (off + bytes + 255) & ~(size_t)255;
        return p;
    };
    __bf16* xbf    = (__bf16*)carve((size_t)ROWS * DM * 2);
    float*  xz     = (float*) carve((size_t)ROWS * 2 * DI * 4);
    float*  xcf    = (float*) carve((size_t)ROWS * DI * 4);
    __bf16* xcb    = (__bf16*)carve((size_t)ROWS * DI * 2);
    float*  dbc    = (float*) carve((size_t)ROWS * 48 * 4);
    float*  dlt    = (float*) carve((size_t)ROWS * DI * 4);
    __bf16* ygb    = (__bf16*)carve((size_t)ROWS * DI * 2);
    __bf16* winb   = (__bf16*)carve((size_t)NLAYERS * 2 * DI * DM * 2);
    __bf16* wxpb   = (__bf16*)carve((size_t)NLAYERS * 48 * DI * 2);
    __bf16* woutb  = (__bf16*)carve((size_t)NLAYERS * DM * DI * 2);

    // x = cond + pe (bf16) ; weights -> bf16
    {
        int n = ROWS * DM;
        prep_kernel<<<(n + 255) / 256, 256, 0, stream>>>(cond, pe, xbf);
        int nw;
        nw = NLAYERS * 2 * DI * DM;
        f2bf_kernel<<<(nw + 255) / 256, 256, 0, stream>>>(w_in, winb, nw);
        nw = NLAYERS * 48 * DI;
        f2bf_kernel<<<(nw + 255) / 256, 256, 0, stream>>>(w_xp, wxpb, nw);
        nw = NLAYERS * DM * DI;
        f2bf_kernel<<<(nw + 255) / 256, 256, 0, stream>>>(w_out, woutb, nw);
    }

    const int nEl = ROWS * DI;
    for (int layer = 0; layer < NLAYERS; ++layer) {
        // in_proj: [32768,256] @ [1024,256]^T -> xz   (CW=16, TDM-staged A)
        {
            int waves = (ROWS / 16) * 16;
            gemm_bf16_wmma<4, 16, DM, true><<<waves / 8, 256, 0, stream>>>(
                xbf, winb + (size_t)layer * 2 * DI * DM, xz, ((__bf16*)0), ROWS);
        }
        // conv + SiLU
        conv_silu_kernel<<<(nEl + 255) / 256, 256, 0, stream>>>(
            xz, conv_w + (size_t)layer * DI * 4, conv_b + (size_t)layer * DI,
            xcf, xcb);
        // x_proj: [32768,512] @ [48,512]^T -> dbc     (CW=1: no A sharing, direct)
        {
            int waves = (ROWS / 16) * 1;
            gemm_bf16_wmma<3, 1, DI, false><<<waves / 8, 256, 0, stream>>>(
                xcb, wxpb + (size_t)layer * 48 * DI, dbc, ((__bf16*)0), ROWS);
        }
        // delta = softplus(dt @ w_dt^T + b)
        delta_kernel<<<(nEl + 255) / 256, 256, 0, stream>>>(
            dbc, w_dt + (size_t)layer * DI * DTR, b_dt + (size_t)layer * DI, dlt);
        // selective scan + gating -> ygate (bf16)
        scan_kernel<<<NB * (DI / 16), 256, 0, stream>>>(
            dbc, dlt, xcf, xz,
            A_log + (size_t)layer * DI * DSTATE, D_skip + (size_t)layer * DI, ygb);
        // out_proj: [32768,512] @ [256,512]^T         (CW=4, TDM-staged A)
        {
            int waves = (ROWS / 16) * 4;
            gemm_bf16_wmma<4, 4, DI, true><<<waves / 8, 256, 0, stream>>>(
                ygb, woutb + (size_t)layer * DM * DI,
                (layer == NLAYERS - 1) ? out : (float*)0, xbf, ROWS);
        }
    }
}
